// Attention_37864431682118
// MI455X (gfx1250) — matrix operand order repaired and checked by reading, not compiled, not run on hardware
//
#include <hip/hip_runtime.h>
#include <hip/hip_bf16.h>

// CDNA5 (gfx1250) fused MHA pipeline, bf16 WMMA everywhere.
// GEMMs: double-buffered LDS via GLOBAL_LOAD_ASYNC_TO_LDS_B128 (ASYNCcnt).
// Flash-attn: S computed transposed (K as A, Q as B) so softmax row stats are
// lane-local; 2 shuffles + 8 alpha broadcasts per 32-key chunk instead of 64.

typedef __attribute__((ext_vector_type(16))) __bf16 v16bf;
typedef __attribute__((ext_vector_type(8)))  float  v8f;
typedef int v4i_vs __attribute__((vector_size(16)));   // matches builtin prototype

union FragAB { uint4 u[2]; v16bf v; };
union Pack8  { __bf16 h[8]; uint4 u; };

#define WMMA_BF16(a, b, c) \
  __builtin_amdgcn_wmma_f32_16x16x32_bf16(false, (a), false, (b), (short)0, (c), false, false)

// ---- async global->LDS (gfx1250) with synchronous fallback ----------------
#if defined(__has_builtin)
#if __has_builtin(__builtin_amdgcn_global_load_async_to_lds_b128) && \
    __has_builtin(__builtin_amdgcn_s_wait_asynccnt)
#define HAVE_ASYNC_LDS 1
#endif
#endif

#ifdef HAVE_ASYNC_LDS
__device__ __forceinline__ void async_cp16(__bf16* lds, const __bf16* g) {
  __builtin_amdgcn_global_load_async_to_lds_b128((v4i_vs*)g, (v4i_vs*)lds, 0, 0);
}
template <int N>
__device__ __forceinline__ void async_wait() {
  __builtin_amdgcn_s_wait_asynccnt(N);
}
#else
__device__ __forceinline__ void async_cp16(__bf16* lds, const __bf16* g) {
  *(uint4*)lds = *(const uint4*)g;
}
template <int N>
__device__ __forceinline__ void async_wait() {}
#endif

// ---------------------------------------------------------------------------
// fp32 -> bf16 elementwise
// ---------------------------------------------------------------------------
__global__ void f32_to_bf16(const float* __restrict__ in, __bf16* __restrict__ out, int n) {
  int i = blockIdx.x * blockDim.x + threadIdx.x;
  for (; i < n; i += gridDim.x * blockDim.x) out[i] = (__bf16)in[i];
}

// W[R][C] fp32 -> Wt[C][R] bf16 (so GEMM B-operand is row-major over K)
__global__ void transpose_f32_to_bf16(const float* __restrict__ in, __bf16* __restrict__ out,
                                      int R, int C) {
  int i = blockIdx.x * blockDim.x + threadIdx.x;
  int n = R * C;
  for (; i < n; i += gridDim.x * blockDim.x) {
    int r = i / C, c = i % C;
    out[(size_t)c * R + r] = (__bf16)in[i];
  }
}

// ---------------------------------------------------------------------------
// Tiled bf16 WMMA GEMM: C[M x 1024] = A[M x 1024] * Bt[1024 x 1024]^T
// Block tile 128x128, K-step 32, 8 waves, wave tile 32x64 -> 8 accumulators.
// MODE 0: out bf16 [b][h][n][dh]   (Q, K)
// MODE 1: out bf16 [b][h][dh][n]   (V transposed; vectorized b128 stores)
// MODE 2: out fp32 [m][1024] + bias (final projection)
// ---------------------------------------------------------------------------
template <int MODE>
__launch_bounds__(256)
__global__ void gemm_bf16_wmma(const __bf16* __restrict__ A, const __bf16* __restrict__ Bt,
                               void* __restrict__ Cout, const float* __restrict__ bias) {
  constexpr int K = 1024, LD = 40;  // padded LDS row stride (80B, 16B aligned)
  __shared__ __bf16 lsA[2][128 * LD];
  __shared__ __bf16 lsB[2][128 * LD];

  const int tid  = threadIdx.x;
  const int w    = tid >> 5;
  const int lane = tid & 31;
  const int lm   = lane & 15;
  const int lh   = lane >> 4;
  const int wm   = (w & 3) * 32;   // wave row base within tile
  const int wn   = (w >> 2) * 64;  // wave col base within tile
  const int mBase = blockIdx.y * 128;
  const int nBase = blockIdx.x * 128;

  const int row0 = tid >> 1, c40 = (tid & 1) * 2;
  const int c41 = c40 + 1;
  const __bf16* gA = A  + (size_t)(mBase)*K;
  const __bf16* gB = Bt + (size_t)(nBase)*K;

  auto issue = [&](int buf, int k0) {
    async_cp16(&lsA[buf][row0 * LD + c40 * 8], &gA[(size_t)row0 * K + k0 + c40 * 8]);
    async_cp16(&lsA[buf][row0 * LD + c41 * 8], &gA[(size_t)row0 * K + k0 + c41 * 8]);
    async_cp16(&lsB[buf][row0 * LD + c40 * 8], &gB[(size_t)row0 * K + k0 + c40 * 8]);
    async_cp16(&lsB[buf][row0 * LD + c41 * 8], &gB[(size_t)row0 * K + k0 + c41 * 8]);
  };

  v8f zero = {0.f, 0.f, 0.f, 0.f, 0.f, 0.f, 0.f, 0.f};
  v8f acc[2][4];
#pragma unroll
  for (int mt = 0; mt < 2; mt++)
#pragma unroll
    for (int nt = 0; nt < 4; nt++) acc[mt][nt] = zero;

  issue(0, 0);  // prologue: fill buffer 0

  for (int k0 = 0; k0 < K; k0 += 32) {
    const int buf = (k0 >> 5) & 1;
    if (k0 + 32 < K) {
      issue(buf ^ 1, k0 + 32);  // stream next tile while computing this one
      async_wait<4>();          // previous tile's copies complete (ours)
    } else {
      async_wait<0>();
    }
    __syncthreads();            // ...and in every other wave

    v16bf af[2], bfrag[4];
#pragma unroll
    for (int mt = 0; mt < 2; mt++) {   // A frag: lane = row, two 16B chunks (ISA A layout)
      FragAB f;
      const __bf16* p = &lsA[buf][(wm + mt * 16 + lm) * LD];
      f.u[0] = *(const uint4*)(p + lh * 8);
      f.u[1] = *(const uint4*)(p + 16 + lh * 8);
      af[mt] = f.v;
    }
#pragma unroll
    for (int nt = 0; nt < 4; nt++) {   // B frag: lane = col, two 16B chunks (same K order as A)
      FragAB f;
      const __bf16* p = &lsB[buf][(wn + nt * 16 + lm) * LD + lh * 8];
      f.u[0] = *(const uint4*)(p);
      f.u[1] = *(const uint4*)(p + 16);
      bfrag[nt] = f.v;
    }
#pragma unroll
    for (int mt = 0; mt < 2; mt++)
#pragma unroll
      for (int nt = 0; nt < 4; nt++)
        acc[mt][nt] = WMMA_BF16(af[mt], bfrag[nt], acc[mt][nt]);
    __syncthreads();  // everyone done reading buf before it is refilled
  }

  // Epilogue. C frag layout: element r -> (m = r + 8*lh, n = lm)
#pragma unroll
  for (int mt = 0; mt < 2; mt++) {
#pragma unroll
    for (int nt = 0; nt < 4; nt++) {
      const int gm0 = mBase + wm + mt * 16 + lh * 8;       // first of 8 consecutive rows
      const int gn  = nBase + wn + nt * 16 + lm;
      if constexpr (MODE == 2) {
        const float bv = bias[gn];
#pragma unroll
        for (int r = 0; r < 8; r++)
          ((float*)Cout)[(size_t)(gm0 + r) * 1024 + gn] = acc[mt][nt][r] + bv;
      } else if constexpr (MODE == 1) {
        // V^T: [b][h][dh][n]; rows r are 8 consecutive nseq -> one b128 store
        const int b = gm0 >> 11, ns = gm0 & 2047;
        const int h = gn >> 6,   dh = gn & 63;
        Pack8 pk;
#pragma unroll
        for (int r = 0; r < 8; r++) pk.h[r] = (__bf16)acc[mt][nt][r];
        *(uint4*)&((__bf16*)Cout)[((size_t)((b * 16 + h) * 64 + dh)) * 2048 + ns] = pk.u;
      } else {
        // Q/K: [b][h][n][dh]
#pragma unroll
        for (int r = 0; r < 8; r++) {
          const int gm = gm0 + r;
          const int b = gm >> 11, ns = gm & 2047;
          const int h = gn >> 6,  dh = gn & 63;
          ((__bf16*)Cout)[((size_t)((b * 16 + h) * 2048 + ns)) * 64 + dh] =
              (__bf16)acc[mt][nt][r];
        }
      }
    }
  }
}

// ---------------------------------------------------------------------------
// Flash attention: grid (16 q-tiles, 64 bh), 256 threads = 8 waves.
// Each wave: 16 q rows, online softmax over key chunks of 32.
// S computed TRANSPOSED: S^T = K_tile(A) x Q(B), so a lane owns a whole query
// row (n = lm = query, elements = keys) -> lane-local max/sum + 1 shfl_xor(16).
// Q,K: bf16 [bh][n][64]; Vt: bf16 [bh][64][n]; O: bf16 [b*n][1024].
// ---------------------------------------------------------------------------
__launch_bounds__(256)
__global__ void flash_attn(const __bf16* __restrict__ Q, const __bf16* __restrict__ Kb,
                           const __bf16* __restrict__ Vt, __bf16* __restrict__ O) {
  constexpr int LDP = 40;
  __shared__ __bf16 lsP[8][16 * LDP];   // wave-private 16x32 P tile, row-major [q][key]

  const int bh = blockIdx.y;
  const int b = bh >> 4, h = bh & 15;
  const int w = threadIdx.x >> 5, lane = threadIdx.x & 31;
  const int lm = lane & 15, lh = lane >> 4;
  const int q0 = blockIdx.x * 128 + w * 16;

  // Q as B-fragments: lane = query column (lm), dh chunks at lh*8 and 16+lh*8, 2 chunks
  v16bf qb[2];
  {
    const __bf16* qrow = Q + ((size_t)bh * 2048 + q0 + lm) * 64;
#pragma unroll
    for (int kc = 0; kc < 2; kc++) {
      FragAB f;
      f.u[0] = *(const uint4*)(qrow + kc * 32 + lh * 8);
      f.u[1] = *(const uint4*)(qrow + kc * 32 + 16 + lh * 8);
      qb[kc] = f.v;
    }
  }

  v8f zero = {0.f, 0.f, 0.f, 0.f, 0.f, 0.f, 0.f, 0.f};
  v8f o[4];
#pragma unroll
  for (int d = 0; d < 4; d++) o[d] = zero;
  float mi = -1e30f, li = 0.f;   // per-lane stats for query lm (dup in both halves)

  const __bf16* kbase = Kb + (size_t)bh * 2048 * 64;
  const __bf16* vbase = Vt + (size_t)bh * 64 * 2048;

  for (int j0 = 0; j0 < 2048; j0 += 32) {
    // S^T tiles: A = K rows (16 keys x 64 dh), B = Q. C: (m=key, n=query)
    v8f st[2];
#pragma unroll
    for (int t = 0; t < 2; t++) {
      v8f a = zero;
      const __bf16* kp = kbase + (size_t)(j0 + t * 16 + lm) * 64;  // lane = key row
#pragma unroll
      for (int kc = 0; kc < 2; kc++) {
        FragAB f;   // A frag of K: two 16B chunks per ISA A layout
        f.u[0] = *(const uint4*)(kp + kc * 32 + lh * 8);
        f.u[1] = *(const uint4*)(kp + kc * 32 + 16 + lh * 8);
        a = WMMA_BF16(f.v, qb[kc], a);
      }
      st[t] = a * 0.125f;   // SCALE = DH^-0.5
    }

    // Lane-local row stats (this lane's query), combine key-halves with 1 shfl
    float rm = fmaxf(st[0][0], st[1][0]);
#pragma unroll
    for (int r = 1; r < 8; r++) rm = fmaxf(rm, fmaxf(st[0][r], st[1][r]));
    rm = fmaxf(rm, __shfl_xor(rm, 16, 32));
    float mn = fmaxf(mi, rm);
    float al = __expf(mi - mn);
    mi = mn;

    float rs = 0.f;
    Pack8 pk[2];
#pragma unroll
    for (int t = 0; t < 2; t++) {
#pragma unroll
      for (int r = 0; r < 8; r++) {
        float p = __expf(st[t][r] - mn);
        rs += p;
        pk[t].h[r] = (__bf16)p;
      }
    }
    rs += __shfl_xor(rs, 16, 32);
    li = li * al + rs;

    // P^T store: lane holds 8 consecutive keys of query lm -> 2 packed b128
    *(uint4*)&lsP[w][lm * LDP + lh * 8]      = pk[0].u;
    *(uint4*)&lsP[w][lm * LDP + 16 + lh * 8] = pk[1].u;

    // Rescale O (C layout m = query = r + 8*lh): broadcast alpha from lane q
    float alr[8];
#pragma unroll
    for (int r = 0; r < 8; r++) alr[r] = __shfl(al, r + 8 * lh, 32);
#pragma unroll
    for (int d = 0; d < 4; d++)
#pragma unroll
      for (int r = 0; r < 8; r++) o[d][r] = o[d][r] * alr[r];

    // O += P * V : P in A layout from LDS, V^T rows are contiguous keys
    FragAB fa;
    const __bf16* pp = &lsP[w][lm * LDP];
    fa.u[0] = *(const uint4*)(pp + lh * 8);
    fa.u[1] = *(const uint4*)(pp + 16 + lh * 8);
#pragma unroll
    for (int d = 0; d < 4; d++) {
      FragAB fv;
      const __bf16* vp = vbase + (size_t)(d * 16 + lm) * 2048 + j0 + lh * 8;
      fv.u[0] = *(const uint4*)vp;
      fv.u[1] = *(const uint4*)(vp + 16);
      o[d] = WMMA_BF16(fa.v, fv.v, o[d]);
    }
  }

  // Normalize (broadcast li per query row) and store O bf16 [b*n][1024]
  float lir[8];
#pragma unroll
  for (int r = 0; r < 8; r++) lir[r] = __shfl(li, r + 8 * lh, 32);
#pragma unroll
  for (int d = 0; d < 4; d++) {
#pragma unroll
    for (int r = 0; r < 8; r++) {
      float val = o[d][r] / lir[r];
      int qr = q0 + r + 8 * lh;
      O[((size_t)(b * 2048 + qr)) * 1024 + h * 64 + d * 16 + lm] = (__bf16)val;
    }
  }
}

// ---------------------------------------------------------------------------
extern "C" void kernel_launch(void* const* d_in, const int* in_sizes, int n_in,
                              void* d_out, int out_size, void* d_ws, size_t ws_size,
                              hipStream_t stream) {
  const float* x  = (const float*)d_in[0];
  const float* Wq = (const float*)d_in[1];
  const float* Wk = (const float*)d_in[2];
  const float* Wv = (const float*)d_in[3];
  const float* Wo = (const float*)d_in[4];
  const float* bo = (const float*)d_in[5];

  char* ws = (char*)d_ws;
  size_t off = 0;
  auto alloc = [&](size_t bytes) -> void* {
    void* p = ws + off;
    off += (bytes + 255) & ~(size_t)255;
    return p;
  };
  __bf16* xb  = (__bf16*)alloc(8192ull * 1024 * 2);
  __bf16* wqt = (__bf16*)alloc(1024ull * 1024 * 2);
  __bf16* wkt = (__bf16*)alloc(1024ull * 1024 * 2);
  __bf16* wvt = (__bf16*)alloc(1024ull * 1024 * 2);
  __bf16* wot = (__bf16*)alloc(1024ull * 1024 * 2);
  __bf16* Qb  = (__bf16*)alloc(8192ull * 1024 * 2);
  __bf16* Kb  = (__bf16*)alloc(8192ull * 1024 * 2);
  __bf16* Vtb = (__bf16*)alloc(8192ull * 1024 * 2);
  __bf16* Ob  = (__bf16*)alloc(8192ull * 1024 * 2);

  f32_to_bf16<<<4096, 256, 0, stream>>>(x, xb, 8192 * 1024);
  transpose_f32_to_bf16<<<2048, 256, 0, stream>>>(Wq, wqt, 1024, 1024);
  transpose_f32_to_bf16<<<2048, 256, 0, stream>>>(Wk, wkt, 1024, 1024);
  transpose_f32_to_bf16<<<2048, 256, 0, stream>>>(Wv, wvt, 1024, 1024);
  transpose_f32_to_bf16<<<2048, 256, 0, stream>>>(Wo, wot, 1024, 1024);

  dim3 g(8, 64);  // Nf/128 x M/128
  gemm_bf16_wmma<0><<<g, 256, 0, stream>>>(xb, wqt, Qb,  nullptr);
  gemm_bf16_wmma<0><<<g, 256, 0, stream>>>(xb, wkt, Kb,  nullptr);
  gemm_bf16_wmma<1><<<g, 256, 0, stream>>>(xb, wvt, Vtb, nullptr);

  flash_attn<<<dim3(16, 64), 256, 0, stream>>>(Qb, Kb, Vtb, Ob);

  gemm_bf16_wmma<2><<<g, 256, 0, stream>>>(Ob, wot, d_out, bo);
}
